// LocalAttention_50483045597888
// MI455X (gfx1250) — matrix operand order repaired and checked by reading, not compiled, not run on hardware
//
#include <hip/hip_runtime.h>
#include <stdint.h>

// ---------------------------------------------------------------------------
// Problem constants (from reference)
// ---------------------------------------------------------------------------
#define BATCH   4
#define SEQ     4096
#define DIM     1024          // DIM_IN == DIM_INNER == DIM_OUT
#define WIN     128
#define NWIN    (SEQ / WIN)   // 32
#define MROWS   (BATCH * SEQ) // 16384

typedef __attribute__((ext_vector_type(16))) __bf16          v16bf;
typedef __attribute__((ext_vector_type(8)))  float           v8f;
typedef __attribute__((ext_vector_type(8)))  unsigned short  v8u16;

// sched_group_barrier masks
#define SG_WMMA   0x008
#define SG_VMEMRD 0x020
#define SG_DSRD   0x100

// ---------------------------------------------------------------------------
// Helpers
// ---------------------------------------------------------------------------
__device__ __forceinline__ unsigned short f32_to_bf16(float f) {
    union { float f; uint32_t u; } c; c.f = f;
    uint32_t u = c.u;
    uint32_t r = (u + 0x7FFFu + ((u >> 16) & 1u)) >> 16;   // RNE
    return (unsigned short)r;
}

__device__ __forceinline__ v8u16 ld8(const unsigned short* p) {
    return *reinterpret_cast<const v8u16*>(p);
}

__device__ __forceinline__ v16bf mk16(v8u16 lo, v8u16 hi) {
    union { v8u16 h[2]; v16bf v; } u;
    u.h[0] = lo; u.h[1] = hi;
    return u.v;
}

__device__ __forceinline__ v8f wmma_bf16(v16bf a, v16bf b, v8f c) {
    return __builtin_amdgcn_wmma_f32_16x16x32_bf16(
        false, a, false, b, (short)0, c, false, false);
}

// gfx1250 async global->LDS copy (ASYNCcnt-tracked).
__device__ __forceinline__ void async_load_b128(uint32_t lds_addr,
                                                uint32_t byte_off,
                                                const void* sbase) {
    asm volatile("global_load_async_to_lds_b128 %0, %1, %2"
                 :
                 : "v"(lds_addr), "v"(byte_off), "s"(sbase)
                 : "memory");
}

__device__ __forceinline__ void wait_async0() {
    asm volatile("s_wait_asynccnt 0x0" ::: "memory");
}

// A-matrix (16x32 bf16) fragment: lane l(0..15) row m0+l,
// K = {k0+h*8..+7} and {k0+h*8+16..+23}, h = lane>>4.
// B-matrix (32x16 bf16) fragment: lane l col n0+l, K = {k0+h*8..+7} and {k0+h*8+16..+23}.

// ---------------------------------------------------------------------------
// Cast / transpose kernels (fp32 -> bf16 bits)
// ---------------------------------------------------------------------------
__global__ void cast_f32_to_bf16(const float* __restrict__ src,
                                 unsigned short* __restrict__ dst, size_t n) {
    size_t i = (size_t)blockIdx.x * blockDim.x + threadIdx.x;
    size_t stride = (size_t)gridDim.x * blockDim.x;
    for (; i < n; i += stride) dst[i] = f32_to_bf16(src[i]);
}

// dst[c*rows + r] = bf16(src[r*srcld + c])
__global__ void transpose_cast_f32_to_bf16(const float* __restrict__ src,
                                           unsigned short* __restrict__ dst,
                                           int rows, int cols, int srcld) {
    size_t n = (size_t)rows * cols;
    size_t i = (size_t)blockIdx.x * blockDim.x + threadIdx.x;
    size_t stride = (size_t)gridDim.x * blockDim.x;
    for (; i < n; i += stride) {
        int c = (int)(i / rows);
        int r = (int)(i - (size_t)c * rows);
        dst[(size_t)c * rows + r] = f32_to_bf16(src[(size_t)r * srcld + c]);
    }
}

// ---------------------------------------------------------------------------
// Shared GEMM core: C(128x128) += A(128xK) * B(K x 128), bf16 WMMA.
// B panel async-staged in LDS (double-buffered); inner chunk schedule pinned
// via sched_group_barrier so DS reads run 2 fragments ahead of the WMMAs.
// ---------------------------------------------------------------------------
#define KCHUNKS (DIM / 32)
#define BPANEL  (128 * 32)    // elements per LDS B panel

template <typename Epilogue>
__device__ __forceinline__ void gemm_core_128x128(
        const unsigned short* __restrict__ A,    // M x DIM, row-major
        const unsigned short* __restrict__ Bt,   // Ncols x DIM (col-major B)
        int wgM, int wgN, Epilogue epi)
{
    __shared__ unsigned short s_b[2 * BPANEL];   // 2 x 8 KB

    const int tid  = threadIdx.x;
    const int wave = tid >> 5;
    const int lane = tid & 31;
    const int l    = lane & 15;
    const int h    = lane >> 4;
    const int m0   = wgM + wave * 16;

    // --- async B staging: thread copies 16 elems (32B) of the 128x32 panel --
    const int copyCol = tid >> 1;             // 0..127
    const int copyKK  = (tid & 1) * 16;       // 0 or 16
    const unsigned short* bsrc =
        Bt + (size_t)(wgN + copyCol) * DIM + copyKK;
    const uint32_t ldsBase =
        (uint32_t)(uintptr_t)&s_b[0] + (uint32_t)(copyCol * 32 + copyKK) * 2;

    auto issueB = [&](int buf, int k0) {
        const uint32_t lds  = ldsBase + (uint32_t)(buf * BPANEL * 2);
        const uint32_t goff = (uint32_t)(k0 * 2);
        async_load_b128(lds,      goff,      bsrc);
        async_load_b128(lds + 16, goff + 16, bsrc);
    };

    const unsigned short* arow = A + (size_t)(m0 + l) * DIM + h * 8;
    auto loadA = [&](int k0) {
        return mk16(ld8(arow + k0), ld8(arow + k0 + 16));
    };
    auto loadBfrag = [&](const unsigned short* bb, int t) {
        const unsigned short* bp = bb + (t * 16 + l) * 32 + h * 8;
        return mk16(ld8(bp), ld8(bp + 16));
    };

    v8f acc[8];
#pragma unroll
    for (int t = 0; t < 8; ++t) acc[t] = (v8f)0.0f;

    // one k-chunk of 8 WMMAs with pinned DS-read pipelining
    auto computeChunk = [&](const unsigned short* bb, v16bf a) {
        v16bf b0 = loadBfrag(bb, 0);
        v16bf b1 = loadBfrag(bb, 1);
#pragma unroll
        for (int t = 0; t < 8; t += 2) {
            acc[t] = wmma_bf16(a, b0, acc[t]);
            if (t + 2 < 8) b0 = loadBfrag(bb, t + 2);
            acc[t + 1] = wmma_bf16(a, b1, acc[t + 1]);
            if (t + 3 < 8) b1 = loadBfrag(bb, t + 3);
        }
        // schedule: 4 DS reads, (1 WMMA, 2 DS reads) x6, 2 WMMAs
        __builtin_amdgcn_sched_group_barrier(SG_DSRD, 4, 0);
#pragma unroll
        for (int t = 0; t < 6; ++t) {
            __builtin_amdgcn_sched_group_barrier(SG_WMMA, 1, 0);
            __builtin_amdgcn_sched_group_barrier(SG_DSRD, 2, 0);
        }
        __builtin_amdgcn_sched_group_barrier(SG_WMMA, 2, 0);
    };

    issueB(0, 0);
    v16bf a_cur = loadA(0);
    wait_async0();
    __syncthreads();

#pragma unroll 1
    for (int kc = 0; kc < KCHUNKS - 1; ++kc) {
        issueB((kc + 1) & 1, (kc + 1) * 32);
        v16bf a_next = loadA((kc + 1) * 32);
        computeChunk(&s_b[0] + (kc & 1) * BPANEL, a_cur);
        a_cur = a_next;
        wait_async0();
        __syncthreads();
    }
    computeChunk(&s_b[0] + ((KCHUNKS - 1) & 1) * BPANEL, a_cur);

    // epilogue: D layout -> lanes 0-15: (M=r, N=l); lanes 16-31: (M=8+r, N=l)
#pragma unroll
    for (int t = 0; t < 8; ++t) {
        const int tbase = wgN + t * 16;        // uniform per tile
        const int col   = tbase + l;
#pragma unroll
        for (int r = 0; r < 8; ++r) {
            const int row = m0 + (h ? 8 + r : r);
            epi(row, col, tbase, acc[t][r]);
        }
    }
}

// ---------------------------------------------------------------------------
// GEMM 1: [Q|K] = x_bf16 (M x 1024) @ Wqk(1024 x 2048, stored transposed).
// Writes q row-major, k row-major, and kT ([b][d][token]).
// ---------------------------------------------------------------------------
__global__ void __launch_bounds__(256)
qk_gemm_kernel(const unsigned short* __restrict__ xb,
               const unsigned short* __restrict__ wqkT,
               unsigned short* __restrict__ q,
               unsigned short* __restrict__ krow,
               unsigned short* __restrict__ kT)
{
    const int wgM = blockIdx.y * 128;
    const int wgN = blockIdx.x * 128;
    gemm_core_128x128(xb, wqkT, wgM, wgN,
        [=](int row, int col, int tbase, float v) {
            const unsigned short bv = f32_to_bf16(v);
            if (tbase < DIM) {                 // uniform (scalar) branch
                q[(size_t)row * DIM + col] = bv;
            } else {
                const int c2  = col - DIM;
                const int bb  = row >> 12;
                const int tok = row & 4095;
                krow[(size_t)row * DIM + c2] = bv;
                kT[((size_t)bb * DIM + c2) * SEQ + tok] = bv;
            }
        });
}

// ---------------------------------------------------------------------------
// GEMM 3: out = attn_out (M x 1024) @ w_out(T) + b_out, fp32 output.
// ---------------------------------------------------------------------------
__global__ void __launch_bounds__(256)
out_gemm_kernel(const unsigned short* __restrict__ ab,
                const unsigned short* __restrict__ woutT,
                const float* __restrict__ bias,
                float* __restrict__ out)
{
    const int wgM = blockIdx.y * 128;
    const int wgN = blockIdx.x * 128;
    gemm_core_128x128(ab, woutT, wgM, wgN,
        [=](int row, int col, int, float v) {
            out[(size_t)row * DIM + col] = v + bias[col];
        });
}

// ---------------------------------------------------------------------------
// Fused windowed attention. One WG = one 32-row strip of one (batch, window).
// grid = (4 strips, 32 windows, 4 batches), block = 256 (8 waves).
// ---------------------------------------------------------------------------
__global__ void __launch_bounds__(256)
local_attn_kernel(const unsigned short* __restrict__ q,     // B*SEQ x 1024
                  const unsigned short* __restrict__ krow,  // B*SEQ x 1024
                  const unsigned short* __restrict__ kT,    // B x 1024 x 4096
                  unsigned short* __restrict__ attn_out)    // B*SEQ x 1024
{
    __shared__ float          s_sim [32][2 * WIN];   // 32 KB
    __shared__ unsigned short s_attn[32][2 * WIN];   // 16 KB

    const int b     = blockIdx.z;
    const int w     = blockIdx.y;
    const int strip = blockIdx.x;
    const int rowBase = strip * 32;

    const int tid  = threadIdx.x;
    const int wave = tid >> 5;
    const int lane = tid & 31;
    const int l    = lane & 15;
    const int h    = lane >> 4;

    const unsigned short* qb  = q    + (size_t)b * SEQ * DIM;
    const unsigned short* kb  = krow + (size_t)b * SEQ * DIM;
    const unsigned short* kTb = kT   + (size_t)b * DIM * SEQ;

    const float scale = 0.03125f;                    // DIM^-0.5

    // ---------------- Phase 1: sim = q @ k2^T (strip: 32 x 256) -------------
#pragma unroll 1
    for (int tt = 0; tt < 4; ++tt) {
        const int t  = wave * 4 + tt;
        const int rt = t >> 4;
        const int ct = t & 15;
        const bool prevBlk = (ct < 8);
        v8f acc = (v8f)0.0f;

        if (!(prevBlk && w == 0)) {
            const int tokc0 = prevBlk ? ((w - 1) * WIN + ct * 16)
                                      : (w * WIN + (ct - 8) * 16);
            const unsigned short* arow =
                qb + (size_t)(w * WIN + rowBase + rt * 16 + l) * DIM + h * 8;
            const unsigned short* bcol =
                kb + (size_t)(tokc0 + l) * DIM + h * 8;

            auto ldA1 = [&](int k0) {
                return mk16(ld8(arow + k0), ld8(arow + k0 + 16));
            };
            auto ldB1 = [&](int k0) {
                return mk16(ld8(bcol + k0), ld8(bcol + k0 + 16));
            };

            v16bf a0 = ldA1(0),  b0 = ldB1(0);
            v16bf a1 = ldA1(32), b1 = ldB1(32);
#pragma unroll 1
            for (int k0 = 0; k0 < DIM - 64; k0 += 64) {
                acc = wmma_bf16(a0, b0, acc);
                a0 = ldA1(k0 + 64); b0 = ldB1(k0 + 64);
                acc = wmma_bf16(a1, b1, acc);
                a1 = ldA1(k0 + 96); b1 = ldB1(k0 + 96);
                __builtin_amdgcn_sched_group_barrier(SG_WMMA,   1, 0);
                __builtin_amdgcn_sched_group_barrier(SG_VMEMRD, 4, 0);
                __builtin_amdgcn_sched_group_barrier(SG_WMMA,   1, 0);
                __builtin_amdgcn_sched_group_barrier(SG_VMEMRD, 4, 0);
            }
            acc = wmma_bf16(a0, b0, acc);      // peeled last chunk pair
            acc = wmma_bf16(a1, b1, acc);
        }
#pragma unroll
        for (int r = 0; r < 8; ++r) {
            const int ri = rt * 16 + (h ? 8 + r : r);
            const int i  = rowBase + ri;
            const int j  = ct * 16 + l;
            float v = acc[r] * scale;
            if (j > i + WIN) v = -3.0e38f;
            s_sim[ri][j] = v;
        }
    }
    __syncthreads();

    // ---------------- Phase 2: row softmax (threads 0..31) ------------------
    if (tid < 32) {
        float mx = -3.4e38f;
        for (int j = 0; j < 2 * WIN; ++j) mx = fmaxf(mx, s_sim[tid][j]);
        float sum = 0.0f;
        for (int j = 0; j < 2 * WIN; ++j) {
            float e = __expf(s_sim[tid][j] - mx);
            s_sim[tid][j] = e;
            sum += e;
        }
        const float rinv = 1.0f / sum;
        for (int j = 0; j < 2 * WIN; ++j)
            s_attn[tid][j] = f32_to_bf16(s_sim[tid][j] * rinv);
    }
    __syncthreads();

    // ---------------- Phase 3: out = attn (32x256) @ v2 (256x1024) ----------
    // v2 == k2 (faithful to the source bug).
    const int jc0 = (w == 0) ? 4 : 0;        // skip zero front block
#pragma unroll 1
    for (int i = 0; i < 16; ++i) {
        const int t  = wave * 16 + i;
        const int rt = t >> 6;
        const int cn = t & 63;
        const int n0 = cn * 16;
        v8f acc = (v8f)0.0f;

        auto ldA3 = [&](int jc) {
            const unsigned short* ar = &s_attn[rt * 16 + l][jc * 32 + h * 8];
            return mk16(ld8(ar), ld8(ar + 16));
        };
        auto ldB3 = [&](int jc) {
            const int tok0 = (jc < 4) ? ((w - 1) * WIN + jc * 32)
                                      : (w * WIN + (jc - 4) * 32);
            const unsigned short* bc =
                kTb + (size_t)(n0 + l) * SEQ + tok0 + h * 8;
            return mk16(ld8(bc), ld8(bc + 16));
        };

        v16bf a0 = ldA3(jc0),     b0 = ldB3(jc0);
        v16bf a1 = ldA3(jc0 + 1), b1 = ldB3(jc0 + 1);
#pragma unroll 1
        for (int jc = jc0; jc + 2 < 8; jc += 2) {
            acc = wmma_bf16(a0, b0, acc);
            a0 = ldA3(jc + 2); b0 = ldB3(jc + 2);
            acc = wmma_bf16(a1, b1, acc);
            a1 = ldA3(jc + 3); b1 = ldB3(jc + 3);
            __builtin_amdgcn_sched_group_barrier(SG_WMMA,   1, 0);
            __builtin_amdgcn_sched_group_barrier(SG_DSRD,   2, 0);
            __builtin_amdgcn_sched_group_barrier(SG_VMEMRD, 2, 0);
            __builtin_amdgcn_sched_group_barrier(SG_WMMA,   1, 0);
            __builtin_amdgcn_sched_group_barrier(SG_DSRD,   2, 0);
            __builtin_amdgcn_sched_group_barrier(SG_VMEMRD, 2, 0);
        }
        acc = wmma_bf16(a0, b0, acc);          // peeled last pair
        acc = wmma_bf16(a1, b1, acc);

#pragma unroll
        for (int r = 0; r < 8; ++r) {
            const int row = w * WIN + rowBase + rt * 16 + (h ? 8 + r : r);
            attn_out[((size_t)b * SEQ + row) * DIM + n0 + l] =
                f32_to_bf16(acc[r]);
        }
    }
}

// ---------------------------------------------------------------------------
// Host-side launcher
// ---------------------------------------------------------------------------
extern "C" void kernel_launch(void* const* d_in, const int* in_sizes, int n_in,
                              void* d_out, int out_size, void* d_ws, size_t ws_size,
                              hipStream_t stream) {
    const float* x     = (const float*)d_in[0];   // B x SEQ x 1024
    const float* w_qkv = (const float*)d_in[1];   // 1024 x 3072
    const float* w_out = (const float*)d_in[2];   // 1024 x 1024
    const float* b_out = (const float*)d_in[3];   // 1024
    float* out = (float*)d_out;

    char* ws = (char*)d_ws;
    unsigned short* xb    = (unsigned short*)(ws);                       // 32 MB
    unsigned short* wqkT  = (unsigned short*)(ws + (size_t)33554432);    //  4 MB
    unsigned short* woutT = (unsigned short*)(ws + (size_t)37748736);    //  2 MB
    unsigned short* qbuf  = (unsigned short*)(ws + (size_t)39845888);    // 32 MB
    unsigned short* krow  = (unsigned short*)(ws + (size_t)73400320);    // 32 MB
    unsigned short* kT    = (unsigned short*)(ws + (size_t)106954752);   // 32 MB
    unsigned short* aout  = (unsigned short*)(ws + (size_t)140509184);   // 32 MB

    cast_f32_to_bf16<<<2048, 256, 0, stream>>>(x, xb, (size_t)MROWS * DIM);
    transpose_cast_f32_to_bf16<<<1024, 256, 0, stream>>>(w_qkv, wqkT,
                                                         DIM, 2 * DIM, 3 * DIM);
    transpose_cast_f32_to_bf16<<<512, 256, 0, stream>>>(w_out, woutT,
                                                        DIM, DIM, DIM);

    qk_gemm_kernel<<<dim3(2 * DIM / 128, MROWS / 128), 256, 0, stream>>>(
        xb, wqkT, qbuf, krow, kT);

    local_attn_kernel<<<dim3(WIN / 32, NWIN, BATCH), 256, 0, stream>>>(
        qbuf, krow, kT, aout);

    out_gemm_kernel<<<dim3(DIM / 128, MROWS / 128), 256, 0, stream>>>(
        aout, woutT, b_out, out);
}
